// MoEFFNBlock_77051713290697
// MI455X (gfx1250) — matrix operand order repaired and checked by reading, not compiled, not run on hardware
//
#include <hip/hip_runtime.h>
#include <hip/hip_bf16.h>

// ---------------------------------------------------------------------------
// MoE FFN block for MI455X (gfx1250, wave32, WMMA).
// Pipeline: pool+LN -> w1/w2 fp32->bf16 -> gate(top2, per-expert lists)
//           -> grouped expert GEMMs (v_wmma_f32_16x16x32_bf16) -> residual.
// Top-2 sparsity: 137 GFLOP of bf16 WMMA, under the ~92us HBM floor from
// reading x (1.07GB) + writing out (1.07GB) at 23.3 TB/s. bf16 weights
// (128 MB) stay resident in the 192 MB L2 while ~256 tile-blocks stream them.
// CDNA5 paths: async global->LDS staging of the gathered token tile,
// TDM tensor_load_to_lds for the gate weights, global prefetch for weights.
// ---------------------------------------------------------------------------

#define B_TOK 4096
#define DIMC  1024
#define HID   4096
#define NE    8
#define MT    32    // tokens per expert tile
#define XPAD  1032  // padded LDS row stride (bf16 elems) for x tile
#define HPAD  72    // padded LDS row stride (bf16 elems) for h tile

typedef __attribute__((ext_vector_type(16))) __bf16 v16bf;
typedef __attribute__((ext_vector_type(8)))  float  v8f;
typedef __attribute__((ext_vector_type(4)))  unsigned int u32x4;
typedef __attribute__((ext_vector_type(8)))  int i32x8;
typedef __attribute__((ext_vector_type(4)))  int i32x4;

union ABv { v16bf v; uint4 u[2]; };

#if __has_builtin(__builtin_amdgcn_tensor_load_to_lds)
#define HAVE_TDM 1
#endif

__device__ __forceinline__ unsigned short f2bf(float f) {
  unsigned int u = __float_as_uint(f);
  u += 0x7FFFu + ((u >> 16) & 1u);   // round-to-nearest-even
  return (unsigned short)(u >> 16);
}
__device__ __forceinline__ float bf2f(unsigned short h) {
  return __uint_as_float(((unsigned int)h) << 16);
}
__device__ __forceinline__ unsigned lds_off_of(const void* p) {
  // low 32 bits of a generic LDS address == byte offset within wave LDS
  return (unsigned)(unsigned long long)p;
}
__device__ __forceinline__ void async_gl_b128(unsigned lds_off, const void* g) {
  asm volatile("global_load_async_to_lds_b128 %0, %1, off"
               :: "v"(lds_off), "v"(g) : "memory");
}
__device__ __forceinline__ void wait_async0() {
  asm volatile("s_wait_asynccnt 0x0" ::: "memory");
}

// ---------------- K0: zero the per-expert counters -------------------------
__global__ void zero_cnt_kernel(int* __restrict__ cnt) {
  if (threadIdx.x < NE) cnt[threadIdx.x] = 0;
}

// ---------------- K1: global-avg-pool + LayerNorm -> bf16 x_norm -----------
__global__ __launch_bounds__(256) void pool_ln_kernel(
    const float* __restrict__ x, const float* __restrict__ gamma,
    const float* __restrict__ beta, unsigned short* __restrict__ xn) {
  int b = blockIdx.x;
  __shared__ float sp[DIMC];
  __shared__ float redA[8], redB[8];
  const float* xb = x + (size_t)b * DIMC * 64;
  int tid = threadIdx.x, lane = tid & 31, wave = tid >> 5;

  float lsum = 0.f, lsq = 0.f;
  #pragma unroll
  for (int ci = 0; ci < 4; ci++) {
    int c = tid + 256 * ci;
    const float4* p = (const float4*)(xb + (size_t)c * 64);
    float s = 0.f;
    #pragma unroll
    for (int q = 0; q < 16; q++) { float4 v = p[q]; s += v.x + v.y + v.z + v.w; }
    float pv = s * (1.0f / 64.0f);
    sp[c] = pv;
    lsum += pv; lsq += pv * pv;
  }
  #pragma unroll
  for (int off = 16; off > 0; off >>= 1) {
    lsum += __shfl_xor(lsum, off);
    lsq  += __shfl_xor(lsq, off);
  }
  if (lane == 0) { redA[wave] = lsum; redB[wave] = lsq; }
  __syncthreads();
  if (tid == 0) {
    float a = 0.f, bb = 0.f;
    #pragma unroll
    for (int i = 0; i < 8; i++) { a += redA[i]; bb += redB[i]; }
    redA[0] = a; redB[0] = bb;
  }
  __syncthreads();
  float mu  = redA[0] * (1.0f / DIMC);
  float var = redB[0] * (1.0f / DIMC) - mu * mu;
  float rstd = rsqrtf(var + 1e-5f);
  #pragma unroll
  for (int ci = 0; ci < 4; ci++) {
    int c = tid + 256 * ci;
    float v = (sp[c] - mu) * rstd * gamma[c] + beta[c];
    xn[(size_t)b * DIMC + c] = f2bf(v);
  }
}

// ---------------- K2: fp32 -> bf16 weight conversion -----------------------
__global__ void cvt_bf16_kernel(const float* __restrict__ src,
                                unsigned short* __restrict__ dst, int n4) {
  int i = blockIdx.x * blockDim.x + threadIdx.x;
  if (i >= n4) return;
  float4 f = ((const float4*)src)[i];
  ushort4 u;
  u.x = f2bf(f.x); u.y = f2bf(f.y); u.z = f2bf(f.z); u.w = f2bf(f.w);
  ((ushort4*)dst)[i] = u;
}

// ---------------- K3: gating (logits, top-2 softmax, list append) ----------
__global__ __launch_bounds__(256) void gate_kernel(
    const unsigned short* __restrict__ xn, const float* __restrict__ wg,
    const float* __restrict__ bg, float* __restrict__ gw,
    int* __restrict__ list, int* __restrict__ cnt) {
  __shared__ __align__(16) float s_wg[NE * DIMC];  // 32 KB staged gate weights
  int tid = threadIdx.x;

#if defined(HAVE_TDM)
  if (tid < 32) {  // wave 0 issues one TDM descriptor: 8 x 1024 fp32 tile
    unsigned long long ga = (unsigned long long)wg;
    u32x4 g0 = { 1u,                                   // count=1, no gather
                 lds_off_of(s_wg),                     // lds_addr
                 (unsigned)ga,                         // global_addr[31:0]
                 (unsigned)((ga >> 32) & 0x1FFFFFFu)   // global_addr[56:32]
                     | (2u << 30) };                   // type=2 (image)
    i32x8 g1 = { (int)(2u << 16),                      // data_size = 4B
                 (int)(((unsigned)DIMC & 0xFFFFu) << 16), // tensor_dim0 lo16
                 (int)((unsigned)NE << 16),            // tensor_dim1 lo16
                 (int)((unsigned)DIMC << 16),          // tile_dim0
                 (int)NE,                              // tile_dim1
                 (int)DIMC,                            // tensor_dim0_stride lo32
                 (int)((unsigned)DIMC << 16),          // tensor_dim1_stride lo16
                 0 };
    i32x4 gz = { 0, 0, 0, 0 };
#if (__clang_major__ >= 23)
    i32x8 gz8 = { 0, 0, 0, 0, 0, 0, 0, 0 };
    __builtin_amdgcn_tensor_load_to_lds(g0, g1, gz, gz, gz8, 0);
#else
    __builtin_amdgcn_tensor_load_to_lds(g0, g1, gz, gz, 0);
#endif
    __builtin_amdgcn_s_wait_tensorcnt(0);
  }
#else
  for (int i = tid; i < NE * DIMC / 4; i += 256) {
    async_gl_b128(lds_off_of(s_wg) + (unsigned)i * 16,
                  (const void*)((const float4*)wg + i));
  }
  wait_async0();
#endif
  __syncthreads();

  int b = blockIdx.x * blockDim.x + tid;
  float logits[NE];
  #pragma unroll
  for (int e = 0; e < NE; e++) logits[e] = bg[e];
  const unsigned short* xr = xn + (size_t)b * DIMC;
  for (int d0 = 0; d0 < DIMC; d0 += 8) {
    uint4 pk = *(const uint4*)(xr + d0);
    unsigned w[4] = { pk.x, pk.y, pk.z, pk.w };
    #pragma unroll
    for (int j = 0; j < 4; j++) {
      float x0 = bf2f((unsigned short)(w[j] & 0xFFFFu));
      float x1 = bf2f((unsigned short)(w[j] >> 16));
      #pragma unroll
      for (int e = 0; e < NE; e++) {
        logits[e] += x0 * s_wg[e * DIMC + d0 + 2 * j];
        logits[e] += x1 * s_wg[e * DIMC + d0 + 2 * j + 1];
      }
    }
  }
  int e0 = 0; float v0 = logits[0];
  #pragma unroll
  for (int e = 1; e < NE; e++) if (logits[e] > v0) { v0 = logits[e]; e0 = e; }
  int e1 = -1; float v1 = -3.0e38f;
  #pragma unroll
  for (int e = 0; e < NE; e++)
    if (e != e0 && logits[e] > v1) { v1 = logits[e]; e1 = e; }
  float bb = __expf(v1 - v0);
  float inv = __builtin_amdgcn_rcpf(1.0f + bb);
  gw[b * 2 + 0] = inv;
  gw[b * 2 + 1] = bb * inv;
  int p0 = atomicAdd(&cnt[e0], 1);
  list[e0 * B_TOK + p0] = (b << 1);
  int p1 = atomicAdd(&cnt[e1], 1);
  list[e1 * B_TOK + p1] = (b << 1) | 1;
}

// ---------------- K4: grouped expert MLP via WMMA bf16 ---------------------
// Block = 8 waves, 32 gathered tokens (async-staged into LDS once).
// Per 64-wide hidden chunk (double-buffered h tile, ONE barrier per chunk):
//   GEMM1: h(32x64) = silu(x_lds(32x1024) @ w1_e^T + b1)  (1 WMMA tile/wave)
//   GEMM2: acc(32x1024) += h(32x64) @ w2_e^T              (32 WMMA/wave)
__global__ __launch_bounds__(256) void expert_ffn_kernel(
    const unsigned short* __restrict__ xn, const unsigned short* __restrict__ w1b,
    const unsigned short* __restrict__ w2b, const float* __restrict__ pb1,
    const float* __restrict__ pb2, const float* __restrict__ gw,
    const int* __restrict__ list, const int* __restrict__ cnt,
    float* __restrict__ ffn) {
  int e = blockIdx.y;
  int n_e = cnt[e];
  int start = blockIdx.x * MT;
  if (start >= n_e) return;
  int valid = n_e - start; if (valid > MT) valid = MT;

  __shared__ __align__(16) unsigned short sh_x[MT * XPAD];     // 32x1024 (+pad)
  __shared__ __align__(16) unsigned short sh_h[2][MT * HPAD];  // double-buffered
  __shared__ int s_ent[MT];
  __shared__ int s_tok[MT];

  int tid = threadIdx.x;
  if (tid < MT) {
    int idx = start + tid; if (idx > n_e - 1) idx = n_e - 1;
    int ent = list[e * B_TOK + idx];
    s_ent[tid] = ent;
    s_tok[tid] = ent >> 1;
  }
  __syncthreads();

  // ---- async gather: 32 token rows of bf16 x_norm -> LDS (64 KB) ----
  {
    unsigned base = lds_off_of(sh_x);
    #pragma unroll 4
    for (int i = tid; i < MT * 128; i += 256) {      // 4096 x 16B chunks
      int row = i >> 7;
      int c16 = i & 127;
      const unsigned short* g = xn + (size_t)s_tok[row] * DIMC + c16 * 8;
      async_gl_b128(base + (unsigned)(row * (XPAD * 2) + c16 * 16), g);
    }
    wait_async0();
  }
  __syncthreads();

  int wave  = tid >> 5;
  int lane  = tid & 31;
  int lhalf = lane & 16;     // 0 or 16: which K-half this lane holds
  int l15   = lane & 15;

  int mt1 = wave & 1;        // GEMM1 tile (mt1, nt1) in the 2x4 grid
  int nt1 = wave >> 1;
  const unsigned short* ax = sh_x + (mt1 * 16 + l15) * XPAD;
  int aoff = lhalf ? 8 : 0;  // A-operand K offset within lane
  int boff = lhalf ? 8 : 0;  // B-operand K offset (0 or 8)
  int nbase2 = wave * 128;   // GEMM2 output column slice for this wave

  v8f vz = {};
  v8f acc[2][8];
  #pragma unroll
  for (int i = 0; i < 2; i++)
    #pragma unroll
    for (int j = 0; j < 8; j++) acc[i][j] = vz;

  for (int hb = 0; hb < HID; hb += 64) {
    unsigned short* hbuf = sh_h[(hb >> 6) & 1];

    // ---- GEMM1: this wave's 16x16 tile of h, K = DIMC ----
    v8f hacc = vz;
    const unsigned short* brow =
        w1b + ((size_t)e * HID + hb + nt1 * 16 + l15) * DIMC;
    if (hb + 64 < HID) __builtin_prefetch(brow + 64 * DIMC, 0, 0);
    #pragma unroll 4
    for (int k0 = 0; k0 < DIMC; k0 += 32) {
      ABv a, bm;
      a.u[0]  = *(const uint4*)(ax + k0 + aoff);        // ds_load_b128
      a.u[1]  = *(const uint4*)(ax + k0 + aoff + 16);
      bm.u[0] = *(const uint4*)(brow + k0 + boff);
      bm.u[1] = *(const uint4*)(brow + k0 + boff + 16);
      hacc = __builtin_amdgcn_wmma_f32_16x16x32_bf16(
          false, a.v, false, bm.v, (short)0, hacc, false, false);
    }
    // bias + silu; store into this chunk's buffer (other buffer may still
    // be read by laggard waves -- safe, different buffer; same buffer's
    // previous readers finished before the previous chunk's barrier)
    {
      int col = nt1 * 16 + l15;
      float bias = pb1[(size_t)e * HID + hb + col];
      int mrow0 = mt1 * 16 + (lhalf ? 8 : 0);
      #pragma unroll
      for (int r = 0; r < 8; r++) {
        float v = hacc[r] + bias;
        float s = v * __builtin_amdgcn_rcpf(1.0f + __expf(-v));  // silu
        hbuf[(mrow0 + r) * HPAD + col] = f2bf(s);
      }
    }
    __syncthreads();   // single barrier per chunk: h stores -> h reads

    // ---- GEMM2: acc(32 x 128-per-wave) += h(32x64) @ w2_e^T(64x128) ----
    #pragma unroll
    for (int kk = 0; kk < 2; kk++) {
      int k0 = kk * 32;
      int kA = k0 + aoff;
      ABv a0, a1;
      a0.u[0] = *(const uint4*)(hbuf + (l15)*HPAD + kA);
      a0.u[1] = *(const uint4*)(hbuf + (l15)*HPAD + kA + 16);
      a1.u[0] = *(const uint4*)(hbuf + (16 + l15) * HPAD + kA);
      a1.u[1] = *(const uint4*)(hbuf + (16 + l15) * HPAD + kA + 16);
      #pragma unroll
      for (int nt = 0; nt < 8; nt++) {
        int d = nbase2 + nt * 16 + l15;
        const unsigned short* wrow =
            w2b + ((size_t)e * DIMC + d) * HID + hb + k0 + boff;
        if (kk == 0 && hb + 64 < HID) __builtin_prefetch(wrow + 64, 0, 0);
        ABv bm;
        bm.u[0] = *(const uint4*)(wrow);
        bm.u[1] = *(const uint4*)(wrow + 16);
        acc[0][nt] = __builtin_amdgcn_wmma_f32_16x16x32_bf16(
            false, a0.v, false, bm.v, (short)0, acc[0][nt], false, false);
        acc[1][nt] = __builtin_amdgcn_wmma_f32_16x16x32_bf16(
            false, a1.v, false, bm.v, (short)0, acc[1][nt], false, false);
      }
    }
  }

  // ---- scatter: ffn[(tok*2+slot)*DIM + d] = (acc + b2) * gate_weight ----
  int mrowL = lhalf ? 8 : 0;
  int   entv[2][8];
  float wv[2][8];
  #pragma unroll
  for (int mt = 0; mt < 2; mt++)
    #pragma unroll
    for (int r = 0; r < 8; r++) {
      int m = mt * 16 + mrowL + r;
      int ent = s_ent[m];
      entv[mt][r] = (m < valid) ? ent : -1;
      wv[mt][r] = gw[ent];
    }
  #pragma unroll
  for (int mt = 0; mt < 2; mt++) {
    #pragma unroll
    for (int nt = 0; nt < 8; nt++) {
      int d = nbase2 + nt * 16 + l15;
      float biasv = pb2[(size_t)e * DIMC + d];
      #pragma unroll
      for (int r = 0; r < 8; r++) {
        if (entv[mt][r] >= 0) {
          ffn[(size_t)entv[mt][r] * DIMC + d] =
              (acc[mt][nt][r] + biasv) * wv[mt][r];
        }
      }
    }
  }
}

// ---------------- K5: residual broadcast ----------------------------------
__global__ __launch_bounds__(256) void residual_kernel(
    const float* __restrict__ x, const float* __restrict__ ffn,
    float* __restrict__ out) {
  int gid = blockIdx.x * 256 + threadIdx.x;        // over B*DIM*16 float4s
  int cidx = gid >> 4;                             // b*DIM + c
  int b = cidx >> 10;
  int c = cidx & 1023;
  float f = ffn[b * 2048 + c] + ffn[b * 2048 + 1024 + c];
  float4 v = ((const float4*)x)[gid];
  v.x += f; v.y += f; v.z += f; v.w += f;
  ((float4*)out)[gid] = v;
}

// ---------------------------------------------------------------------------
extern "C" void kernel_launch(void* const* d_in, const int* in_sizes, int n_in,
                              void* d_out, int out_size, void* d_ws, size_t ws_size,
                              hipStream_t stream) {
  const float* x     = (const float*)d_in[0];
  const float* gamma = (const float*)d_in[1];
  const float* beta  = (const float*)d_in[2];
  const float* wg    = (const float*)d_in[3];
  const float* bg    = (const float*)d_in[4];
  const float* w1    = (const float*)d_in[5];
  const float* b1    = (const float*)d_in[6];
  const float* w2    = (const float*)d_in[7];
  const float* b2    = (const float*)d_in[8];
  float* out = (float*)d_out;

  size_t off = 0;
  char* base = (char*)d_ws;
  auto carve = [&](size_t bytes) -> void* {
    void* p = base + off;
    off += (bytes + 255) & ~(size_t)255;
    return p;
  };
  unsigned short* xn  = (unsigned short*)carve((size_t)B_TOK * DIMC * 2);
  unsigned short* w1b = (unsigned short*)carve((size_t)NE * HID * DIMC * 2);
  unsigned short* w2b = (unsigned short*)carve((size_t)NE * DIMC * HID * 2);
  float* ffn = (float*)carve((size_t)B_TOK * 2 * DIMC * 4);
  float* gw  = (float*)carve((size_t)B_TOK * 2 * 4);
  int* list  = (int*)carve((size_t)NE * B_TOK * 4);
  int* cnt   = (int*)carve(256);

  zero_cnt_kernel<<<1, 32, 0, stream>>>(cnt);
  pool_ln_kernel<<<B_TOK, 256, 0, stream>>>(x, gamma, beta, xn);

  int n4 = NE * HID * DIMC / 4;
  cvt_bf16_kernel<<<(n4 + 255) / 256, 256, 0, stream>>>(w1, w1b, n4);
  cvt_bf16_kernel<<<(n4 + 255) / 256, 256, 0, stream>>>(w2, w2b, n4);

  gate_kernel<<<B_TOK / 256, 256, 0, stream>>>(xn, wg, bg, gw, list, cnt);

  expert_ffn_kernel<<<dim3(B_TOK / MT, NE), 256, 0, stream>>>(
      xn, w1b, w2b, b1, b2, gw, list, cnt, ffn);

  residual_kernel<<<(B_TOK * DIMC * 16) / 256, 256, 0, stream>>>(x, ffn, out);
}
